// SchNetInteraction_84052509983251
// MI455X (gfx1250) — matrix operand order repaired and checked by reading, compiled, not run on hardware
//
#include <hip/hip_runtime.h>
#include <hip/hip_bf16.h>

// ---------------------------------------------------------------------------
// SchNet interaction block, fused for MI455X (gfx1250, wave32, WMMA bf16).
//
// Shapes: C=32 configs, A=128 atoms, B=128 basis, G=64 gaussians, F=128 filters.
// Heavy path: w = tanh(f_ij@W1+b1)@W2+b2 fused with the neighbor
// gather-multiply-reduce, so the 266MB "w" tensor never touches HBM.
// HBM traffic ~= f_ij (133 MB) -> ~5.7us floor at 23.3 TB/s.
// tanh is a branch-free v_exp/v_rcp sequence (libm tanhf's divergent path
// would dominate over the 66M activations of the filter network).
// ---------------------------------------------------------------------------

typedef __bf16 bf16_t;
typedef bf16_t v16bf __attribute__((ext_vector_type(16)));
typedef bf16_t v8bf  __attribute__((ext_vector_type(8)));
typedef float  v8f   __attribute__((ext_vector_type(8)));

__device__ __forceinline__ v8f wmma_bf16(v16bf a, v16bf b, v8f c) {
  // v_wmma_f32_16x16x32_bf16  D = A(16x32) * B(32x16) + C(16x16 f32)
  return __builtin_amdgcn_wmma_f32_16x16x32_bf16(false, a, false, b, (short)0, c,
                                                 false, false);
}

// Branch-free tanh: tanh(x) = 1 - 2/(exp2(2*log2e*x) + 1).
// v_exp_f32 is native exp2; v_rcp_f32 avoids the IEEE-div expansion.
// Clamp keeps exp2 finite; |x|>=15 saturates to +-1 exactly as tanh does.
__device__ __forceinline__ float fast_tanh(float x) {
  float xc = fminf(fmaxf(x, -15.0f), 15.0f);
  float e  = __builtin_amdgcn_exp2f(xc * 2.885390082f);  // 2*log2(e)
  float r  = __builtin_amdgcn_rcpf(e + 1.0f);
  return fmaf(-2.0f, r, 1.0f);
}

// A-fragment (16x32 bf16), source row-major with row stride ldk halves.
// Contiguous k order (k = 16h + s), not AMD's table: lane m (0..15) holds row m, K = kb..kb+15,
// kb = 16*(lane>=16). Two contiguous 16B chunks per lane.
__device__ __forceinline__ v16bf load_a_frag(const bf16_t* base, int lane,
                                             int row0, int k0, int ldk) {
  int m  = lane & 15;
  int kb = (lane >> 4) << 4;
  const bf16_t* p = base + (row0 + m) * ldk + k0 + kb;
  v8bf lo = *(const v8bf*)p;
  v8bf hi = *(const v8bf*)(p + 8);
  return __builtin_shufflevector(lo, hi, 0, 1, 2, 3, 4, 5, 6, 7, 8, 9, 10, 11,
                                 12, 13, 14, 15);
}

// B-fragment (32x16 bf16), source column-major: colmaj[n*ldk + k].
// Contiguous k order, same as A: lane n (0..15) holds column n, halves = K k0..k0+15 (lanes<16)
// or k0+16..k0+31 (lanes>=16). One contiguous 32B run per lane.
__device__ __forceinline__ v16bf load_b_frag(const bf16_t* colmaj, int lane,
                                             int n0, int k0, int ldk) {
  int n  = lane & 15;
  int ko = (lane >> 4) << 4;
  const bf16_t* p = colmaj + (n0 + n) * ldk + k0 + ko;
  v8bf lo = *(const v8bf*)p;
  v8bf hi = *(const v8bf*)(p + 8);
  return __builtin_shufflevector(lo, hi, 0, 1, 2, 3, 4, 5, 6, 7, 8, 9, 10, 11,
                                 12, 13, 14, 15);
}

// ---------------------------------------------------------------------------
// Prep: fp32 row-major [K][N]  ->  bf16 column-major [N][K]
// ---------------------------------------------------------------------------
__global__ void transpose_cvt_kernel(const float* __restrict__ src,
                                     bf16_t* __restrict__ dst, int K, int N) {
  int idx = blockIdx.x * blockDim.x + threadIdx.x;
  if (idx >= K * N) return;
  int n = idx / K, k = idx - n * K;
  dst[idx] = (bf16_t)src[k * N + n];
}

// ---------------------------------------------------------------------------
// Generic M x 128 GEMM, K=128: C = act(A @ B + bias). A fp32 row-major,
// B bf16 column-major. One WG per 128 rows; wave wv owns column strip
// [16wv,16wv+16) so each B fragment is loaded exactly once per WG (L2 hit).
// ---------------------------------------------------------------------------
#define LDA_PAD 136  // 128+8 halves: 16B loads land on distinct bank groups

__global__ __launch_bounds__(256) void gemm128_kernel(
    const float* __restrict__ A, const bf16_t* __restrict__ Bt,
    const float* __restrict__ bias, float* __restrict__ C, int act) {
  __shared__ __align__(16) bf16_t sA[128 * LDA_PAD];
  const int tid = threadIdx.x;
  const int rb  = blockIdx.x;
  const float* Ab = A + (size_t)rb * 128 * 128;

  // stage A tile (fp32 -> bf16), float4 loads for b128 traffic
  for (int q = tid; q < 128 * 32; q += 256) {
    int r = q >> 5, k4 = (q & 31) * 4;
    float4 f4 = ((const float4*)Ab)[q];
    bf16_t* d = &sA[r * LDA_PAD + k4];
    d[0] = (bf16_t)f4.x; d[1] = (bf16_t)f4.y;
    d[2] = (bf16_t)f4.z; d[3] = (bf16_t)f4.w;
  }
  __syncthreads();

  const int wv = tid >> 5, lane = tid & 31;
  const int n_ = lane & 15, hi = lane >> 4;
  const int n0 = wv * 16;

  v16bf bfr[4];
#pragma unroll
  for (int kc = 0; kc < 4; ++kc) bfr[kc] = load_b_frag(Bt, lane, n0, kc * 32, 128);
  const float bv = bias[n0 + n_];

#pragma unroll
  for (int mt = 0; mt < 8; ++mt) {
    v8f acc = {};
#pragma unroll
    for (int kc = 0; kc < 4; ++kc)
      acc = wmma_bf16(load_a_frag(sA, lane, mt * 16, kc * 32, LDA_PAD), bfr[kc], acc);
#pragma unroll
    for (int r = 0; r < 8; ++r) {
      float o = acc[r] + bv;
      if (act) o = fast_tanh(o);
      C[(size_t)(rb * 128 + mt * 16 + r + 8 * hi) * 128 + n0 + n_] = o;
    }
  }
}

// ---------------------------------------------------------------------------
// Fused filter network + neighbor reduce. One WG per (config c, atom i):
//   h = tanh(f_ij[c,i] @ W1 + b1)            (127x64 @ 64x128)
//   w = h @ W2 + b2                          (127x128 @ 128x128)
//   v[c,i,f] = sum_j w[j,f] * y[c, nbh(i,j), f]   (fused, w stays on-chip)
// ---------------------------------------------------------------------------
#define LDF_PAD 72   // 64+8
#define LDH_PAD 136  // 128+8

__global__ __launch_bounds__(256) void schnet_filter_kernel(
    const float* __restrict__ f_ij, const bf16_t* __restrict__ W1t,
    const bf16_t* __restrict__ W2t, const float* __restrict__ b1,
    const float* __restrict__ b2, const float* __restrict__ y,
    float* __restrict__ v) {
  __shared__ __align__(16) bf16_t sF[128 * LDF_PAD];  // 18 KB
  __shared__ __align__(16) bf16_t sH[128 * LDH_PAD];  // 34 KB

  const int tid = threadIdx.x;
  const int blk = blockIdx.x;
  const int c = blk >> 7, i = blk & 127;
  const float* fb = f_ij + (size_t)(c * 128 + i) * 127 * 64;

  // stage f tile (127x64 fp32 -> bf16), pad row 127 with zeros
  for (int q = tid; q < 127 * 16; q += 256) {
    int j = q >> 4, k4 = (q & 15) * 4;
    float4 f4 = ((const float4*)fb)[q];
    bf16_t* d = &sF[j * LDF_PAD + k4];
    d[0] = (bf16_t)f4.x; d[1] = (bf16_t)f4.y;
    d[2] = (bf16_t)f4.z; d[3] = (bf16_t)f4.w;
  }
  for (int k = tid; k < 64; k += 256) sF[127 * LDF_PAD + k] = (bf16_t)0.0f;
  __syncthreads();

  const int wv = tid >> 5, lane = tid & 31;
  const int n_ = lane & 15, hi = lane >> 4;
  const int n0 = wv * 16;

  // ---- GEMM1: h = tanh(f @ W1 + b1), column strip n0..n0+15 ----
  {
    v16bf bf0 = load_b_frag(W1t, lane, n0, 0, 64);
    v16bf bf1 = load_b_frag(W1t, lane, n0, 32, 64);
    const float bias1 = b1[n0 + n_];
#pragma unroll
    for (int mt = 0; mt < 8; ++mt) {
      v8f acc = {};
      acc = wmma_bf16(load_a_frag(sF, lane, mt * 16, 0, LDF_PAD), bf0, acc);
      acc = wmma_bf16(load_a_frag(sF, lane, mt * 16, 32, LDF_PAD), bf1, acc);
#pragma unroll
      for (int r = 0; r < 8; ++r) {
        int j = mt * 16 + r + 8 * hi;
        sH[j * LDH_PAD + n0 + n_] = (bf16_t)fast_tanh(acc[r] + bias1);
      }
    }
  }
  __syncthreads();

  // ---- GEMM2 + gather-multiply-reduce ----
  {
    v16bf bfr[4];
#pragma unroll
    for (int kc = 0; kc < 4; ++kc)
      bfr[kc] = load_b_frag(W2t, lane, n0, kc * 32, 128);
    const float bias2 = b2[n0 + n_];
    const float* yc = y + (size_t)c * 128 * 128;

    float partial = 0.0f;
#pragma unroll
    for (int mt = 0; mt < 8; ++mt) {
      v8f acc = {};
#pragma unroll
      for (int kc = 0; kc < 4; ++kc)
        acc = wmma_bf16(load_a_frag(sH, lane, mt * 16, kc * 32, LDH_PAD), bfr[kc], acc);
#pragma unroll
      for (int r = 0; r < 8; ++r) {
        int j = mt * 16 + r + 8 * hi;
        if (j < 127) {
          int nb = (j < i) ? j : j + 1;  // neighbor index, skipping self
          partial += (acc[r] + bias2) * yc[nb * 128 + n0 + n_];
        }
      }
    }
    // lanes l and l+16 hold the same column: one butterfly finishes the sum
    partial += __shfl_xor(partial, 16, 32);
    if (hi == 0) v[(size_t)(c * 128 + i) * 128 + n0 + n_] = partial;
  }
}

// ---------------------------------------------------------------------------
// Host-side launch
// ---------------------------------------------------------------------------
extern "C" void kernel_launch(void* const* d_in, const int* in_sizes, int n_in,
                              void* d_out, int out_size, void* d_ws, size_t ws_size,
                              hipStream_t stream) {
  (void)in_sizes; (void)n_in; (void)out_size; (void)ws_size;
  const float* x       = (const float*)d_in[0];   // (32,128,128)
  const float* f_ij    = (const float*)d_in[1];   // (32,128,127,64)
  const float* W1      = (const float*)d_in[2];   // (64,128)
  const float* b1      = (const float*)d_in[3];   // (128)
  const float* W2      = (const float*)d_in[4];   // (128,128)
  const float* b2      = (const float*)d_in[5];   // (128)
  const float* W_in2f  = (const float*)d_in[6];   // (128,128)
  const float* b_in2f  = (const float*)d_in[7];   // (128)
  const float* W_f2out = (const float*)d_in[8];   // (128,128)
  const float* b_f2out = (const float*)d_in[9];   // (128)
  const float* W_dense = (const float*)d_in[10];  // (128,128)
  const float* b_dense = (const float*)d_in[11];  // (128)
  float* out = (float*)d_out;                     // (32,128,128)

  char* ws = (char*)d_ws;
  // bf16 transposed weights
  bf16_t* w1t  = (bf16_t*)(ws + 0);        //  64*128 bf16 = 16 KB
  bf16_t* w2t  = (bf16_t*)(ws + 16384);    // 128*128 bf16 = 32 KB
  bf16_t* wi2f = (bf16_t*)(ws + 49152);    // 32 KB
  bf16_t* wf2o = (bf16_t*)(ws + 81920);    // 32 KB
  bf16_t* wden = (bf16_t*)(ws + 114688);   // 32 KB
  // fp32 intermediates (each 4096x128 = 2 MB)
  float* y_ws = (float*)(ws + 147456);
  float* v_ws = (float*)(ws + 2244608);
  float* t_ws = (float*)(ws + 4341760);
  // total ws usage: 6,438,912 bytes (~6.2 MB)

  // 1) weight prep: transpose + fp32->bf16 (all tiny, L2-resident afterwards)
  transpose_cvt_kernel<<<(64 * 128 + 255) / 256, 256, 0, stream>>>(W1, w1t, 64, 128);
  transpose_cvt_kernel<<<(128 * 128 + 255) / 256, 256, 0, stream>>>(W2, w2t, 128, 128);
  transpose_cvt_kernel<<<(128 * 128 + 255) / 256, 256, 0, stream>>>(W_in2f, wi2f, 128, 128);
  transpose_cvt_kernel<<<(128 * 128 + 255) / 256, 256, 0, stream>>>(W_f2out, wf2o, 128, 128);
  transpose_cvt_kernel<<<(128 * 128 + 255) / 256, 256, 0, stream>>>(W_dense, wden, 128, 128);

  // 2) y = x @ W_in2f + b_in2f          (4096x128, stays in L2)
  gemm128_kernel<<<32, 256, 0, stream>>>(x, wi2f, b_in2f, y_ws, 0);

  // 3) fused filter net + neighbor reduce -> v  (the heavy, HBM-bound part)
  schnet_filter_kernel<<<32 * 128, 256, 0, stream>>>(f_ij, w1t, w2t, b1, b2,
                                                     y_ws, v_ws);

  // 4) t = tanh(v @ W_f2out + b_f2out)
  gemm128_kernel<<<32, 256, 0, stream>>>(v_ws, wf2o, b_f2out, t_ws, 1);

  // 5) out = t @ W_dense + b_dense
  gemm128_kernel<<<32, 256, 0, stream>>>(t_ws, wden, b_dense, out, 0);
}
